// Camera2BEV_10857677324979
// MI455X (gfx1250) — matrix operand order repaired and checked by reading, not compiled, not run on hardware
//
#include <hip/hip_runtime.h>
#include <cstdint>
#include <cstddef>

typedef __bf16 bf16_t;
typedef __attribute__((ext_vector_type(8)))  __bf16 v8bf;
typedef __attribute__((ext_vector_type(16))) __bf16 v16bf;
typedef __attribute__((ext_vector_type(8)))  float  v8f;
typedef __attribute__((ext_vector_type(4)))  unsigned int u32x4;
typedef __attribute__((ext_vector_type(4)))  int i32x4;
typedef __attribute__((ext_vector_type(8)))  int i32x8;

#define Q_N    16384
#define M_N    32768      // Q * bs  (also bs*128*128 pixels)
#define RHO_H  130        // 128 + 2 halo columns (zero padded)
#define PI_F   3.14159265358979323846f

#define ACT_NONE    0
#define ACT_SIGMOID 2

#if __has_builtin(__builtin_amdgcn_tensor_load_to_lds) && __has_builtin(__builtin_amdgcn_s_wait_tensorcnt)
#define USE_TDM 1
#else
#define USE_TDM 0
#endif

__device__ __forceinline__ float sigm(float x){ return 1.f/(1.f+expf(-x)); }
__device__ __forceinline__ float inv_sigm(float x){
  x = fminf(fmaxf(x, 0.f), 1.f);
  return logf(fmaxf(x, 1e-5f) / fmaxf(1.f - x, 1e-5f));
}
__device__ __forceinline__ v16bf join16(v8bf a, v8bf b){
  return __builtin_shufflevector(a, b, 0,1,2,3,4,5,6,7,8,9,10,11,12,13,14,15);
}

// TDM: load a (rows=16) x (cols=256 bf16) tile, row stride 2304 elements, into LDS.
__device__ __forceinline__ void tdm_load_wtile(const bf16_t* gsrc, unsigned lds_addr){
#if USE_TDM
  unsigned long long ga = (unsigned long long)(uintptr_t)gsrc;
  u32x4 g0;
  g0[0] = 1u;                                   // count=1, user mode
  g0[1] = lds_addr;                             // lds_addr (bytes)
  g0[2] = (unsigned)(ga & 0xffffffffu);         // global_addr[31:0]
  g0[3] = (unsigned)((ga >> 32) & 0x01ffffffu)  // global_addr[56:32]
        | (2u << 30);                           // type=2 (image)
  i32x8 g1;
  g1[0] = (1 << 16);          // wg_mask=0 | data_size=1 (2 bytes)
  g1[1] = (256 << 16);        // atomic_barrier_addr=0 | tensor_dim0[15:0]=256
  g1[2] = (16 << 16);         // tensor_dim0[31:16]=0 | tensor_dim1[15:0]=16
  g1[3] = (256 << 16);        // tensor_dim1[31:16]=0 | tile_dim0=256
  g1[4] = 16;                 // tile_dim1=16 | tile_dim2=0
  g1[5] = 2304;               // tensor_dim0_stride[31:0]
  g1[6] = 0;                  // stride hi | tensor_dim1_stride lo
  g1[7] = 0;
  i32x4 g2 = {0,0,0,0}, g3 = {0,0,0,0};
#if __clang_major__ >= 23
  i32x8 g4 = {0,0,0,0,0,0,0,0};
  __builtin_amdgcn_tensor_load_to_lds(g0, g1, g2, g3, g4, 0);
#else
  __builtin_amdgcn_tensor_load_to_lds(g0, g1, g2, g3, 0);
#endif
#else
  (void)gsrc; (void)lds_addr;
#endif
}

// ---------------------------------------------------------------- utility
__global__ void zero_f32(float* __restrict__ p, int n){
  int i = blockIdx.x * blockDim.x + threadIdx.x;
  if (i < n) p[i] = 0.f;
}
__global__ void f32_to_bf16(const float* __restrict__ in, bf16_t* __restrict__ out, int n){
  int i = blockIdx.x * blockDim.x + threadIdx.x;
  if (i < n) out[i] = (bf16_t)in[i];
}
// conv weight (co,ci,3,3) -> (co, tap, ci) bf16
__global__ void conv_w_repack(const float* __restrict__ W, bf16_t* __restrict__ Wr){
  int idx = blockIdx.x * blockDim.x + threadIdx.x;   // co*2304 + tap*256 + ci
  if (idx >= 256*2304) return;
  int co = idx / 2304, rem = idx % 2304;
  int tap = rem >> 8, ci = rem & 255;
  Wr[idx] = (bf16_t)W[(co*256 + ci)*9 + tap];
}
// zero the rho halo columns of a (bs,128,130,256) bf16 buffer
__global__ void halo_zero(bf16_t* __restrict__ In){
  int i = blockIdx.x * blockDim.x + threadIdx.x;     // 2*128*2*256
  if (i >= 2*128*2*256) return;
  int c = i & 255;
  int t = i >> 8;               // 0..511
  int side = t & 1;
  int row  = t >> 1;            // b*128+phi, 0..255
  In[((size_t)row * RHO_H + (side ? (RHO_H-1) : 0)) * 256 + c] = (bf16_t)0.f;
}

// ---------------------------------------------------------------- init
__global__ __launch_bounds__(256)
void init_kernel(const float* __restrict__ phi_query, const float* __restrict__ rho_query,
                 const float* __restrict__ ref_W, const float* __restrict__ ref_b,
                 const float* __restrict__ zz,
                 float* __restrict__ qe, float* __restrict__ outF, float* __restrict__ zzb)
{
  int q = blockIdx.x, c = threadIdx.x;
  int p = q >> 7, r = q & 127;
  float lo = phi_query[p*512 + c]       + rho_query[r*512 + c];
  float hi = phi_query[p*512 + 256 + c] + rho_query[r*512 + 256 + c];
  size_t m0 = (size_t)(q*2) * 256 + c;
  qe[m0] = lo;  qe[m0 + 256] = lo;
  outF[m0] = hi; outF[m0 + 256] = hi;
  __shared__ float red[256];
  red[c] = lo * ref_W[c];
  __syncthreads();
  for (int s = 128; s > 0; s >>= 1){ if (c < s) red[c] += red[c + s]; __syncthreads(); }
  if (c == 0) {
    float rp = red[0] + ref_b[0];
    float v  = sigm(inv_sigm(zz[q]) + rp);
    zzb[q] = v; zzb[Q_N + q] = v;
  }
}

// ---------------------------------------------------------------- fused attn + sampling + agg
__global__ __launch_bounds__(256)
void sample_agg(const float* __restrict__ x, const float* __restrict__ l2i,
                const float* __restrict__ outF, const float* __restrict__ qe,
                const float* __restrict__ zzb,
                const float* __restrict__ awW, const float* __restrict__ awb,
                bf16_t* __restrict__ aggB)
{
  int m = blockIdx.x, c = threadIdx.x;
  int q = m >> 1, b = m & 1;
  __shared__ float qt[256];
  __shared__ float attn24[24];
  qt[c] = outF[(size_t)m*256 + c] + qe[(size_t)m*256 + c];
  __syncthreads();
  if (c < 24) {
    float d = awb[c];
    const float* wr = awW + c*256;
    for (int e = 0; e < 256; ++e) d += qt[e] * wr[e];
    attn24[c] = sigm(d);
  }
  __syncthreads();

  float rq = (float)(q & 127) * (50.f/128.f) + (25.f/128.f);
  float pq = (float)(q >> 7) * (2.f*PI_F/128.f) - PI_F + (PI_F/128.f);
  float cx = rq * cosf(pq), cy = rq * sinf(pq);
  float zv = zzb[b*Q_N + q] * 8.f - 5.f;

  float acc = 0.f;
  for (int n = 0; n < 6; ++n) {
    const float* Lm = l2i + (size_t)(b*6 + n)*16;
    float c0 = Lm[0]*cx + Lm[1]*cy + Lm[2]*zv  + Lm[3];
    float c1 = Lm[4]*cx + Lm[5]*cy + Lm[6]*zv  + Lm[7];
    float c2 = Lm[8]*cx + Lm[9]*cy + Lm[10]*zv + Lm[11];
    float denom = fmaxf(c2, 1e-5f);
    float gx = (c0/denom * (1.f/1408.f) - 0.5f) * 2.f;
    float gy = (c1/denom * (1.f/512.f)  - 0.5f) * 2.f;
    bool ms = (c2 > 1e-5f) && gx > -1.f && gx < 1.f && gy > -1.f && gy < 1.f;
    if (!ms) continue;
    float wn = attn24[n*4] + attn24[n*4+1] + attn24[n*4+2] + attn24[n*4+3];
    float gxp = (gx + 1.f) * 0.5f * 88.f - 0.5f;
    float gyp = (gy + 1.f) * 0.5f * 32.f - 0.5f;
    float x0f = floorf(gxp), y0f = floorf(gyp);
    float wx1 = gxp - x0f,  wy1 = gyp - y0f;
    int x0 = (int)x0f, y0 = (int)y0f;
    const float* fb = x + (size_t)(b*6 + n) * 32 * 88 * 256;
    auto tap = [&](int xi, int yi, float w){
      if (xi >= 0 && xi < 88 && yi >= 0 && yi < 32)
        acc += wn * w * fb[((size_t)yi*88 + xi)*256 + c];
    };
    tap(x0,   y0,   (1.f-wx1)*(1.f-wy1));
    tap(x0+1, y0,   wx1      *(1.f-wy1));
    tap(x0,   y0+1, (1.f-wx1)*wy1);
    tap(x0+1, y0+1, wx1      *wy1);
  }
  aggB[(size_t)m*256 + c] = (bf16_t)acc;
}

// ---------------------------------------------------------------- wide bf16 WMMA GEMM (16M x 64N per wave)
// C = A[M,K] @ Bw[N,K]^T + bias (+Res). grid = (M/64, N/64), block 128.
template<bool HAS_RES, bool WF32, bool CONVIN>
__global__ __launch_bounds__(128)
void gemm_wide(const bf16_t* __restrict__ A, const bf16_t* __restrict__ Bw,
               const float* __restrict__ bias, const float* __restrict__ Res,
               float* __restrict__ Cf, bf16_t* __restrict__ Cb,
               int M, int N, int K)
{
  const int lane = threadIdx.x & 31;
  const int wave = threadIdx.x >> 5;
  const int half = lane >> 4;
  const int l15  = lane & 15;
  const int m0 = (blockIdx.x * 4 + wave) * 16;
  const int n0 = blockIdx.y * 64;

  v8f acc0 = {}, acc1 = {}, acc2 = {}, acc3 = {};
  const bf16_t* ap  = A  + (size_t)(m0 + l15) * K + half * 16;
  const bf16_t* bp0 = Bw + (size_t)(n0 + l15) * K + half * 16;
  const bf16_t* bp1 = bp0 + (size_t)16 * K;
  const bf16_t* bp2 = bp0 + (size_t)32 * K;
  const bf16_t* bp3 = bp0 + (size_t)48 * K;
  for (int k0 = 0; k0 < K; k0 += 32) {
    v16bf af = join16(*(const v8bf*)(ap + k0), *(const v8bf*)(ap + k0 + 8));
    v16bf b0 = join16(*(const v8bf*)(bp0 + k0), *(const v8bf*)(bp0 + k0 + 8));
    v16bf b1 = join16(*(const v8bf*)(bp1 + k0), *(const v8bf*)(bp1 + k0 + 8));
    v16bf b2 = join16(*(const v8bf*)(bp2 + k0), *(const v8bf*)(bp2 + k0 + 8));
    v16bf b3 = join16(*(const v8bf*)(bp3 + k0), *(const v8bf*)(bp3 + k0 + 8));
    acc0 = __builtin_amdgcn_wmma_f32_16x16x32_bf16(false, af, false, b0, (short)0, acc0, false, false);
    acc1 = __builtin_amdgcn_wmma_f32_16x16x32_bf16(false, af, false, b1, (short)0, acc1, false, false);
    acc2 = __builtin_amdgcn_wmma_f32_16x16x32_bf16(false, af, false, b2, (short)0, acc2, false, false);
    acc3 = __builtin_amdgcn_wmma_f32_16x16x32_bf16(false, af, false, b3, (short)0, acc3, false, false);
  }
  auto epi = [&](v8f a, int nt){
    const int nn = n0 + nt*16 + l15;
#pragma unroll
    for (int r = 0; r < 8; ++r) {
      int mm = m0 + r + half * 8;
      float v = a[r] + bias[nn];
      if constexpr (HAS_RES) v += Res[(size_t)mm * N + nn];
      if constexpr (WF32)    Cf[(size_t)mm * N + nn] = v;
      if constexpr (CONVIN) {                       // (b, phi, rho+1, c) halo channel-last
        int qq = mm >> 1, bb = mm & 1;
        size_t row = (size_t)(bb*128 + (qq >> 7));
        Cb[(row * RHO_H + (qq & 127) + 1) * 256 + nn] = (bf16_t)v;
      }
    }
  };
  epi(acc0, 0); epi(acc1, 1); epi(acc2, 2); epi(acc3, 3);
}

// ---------------------------------------------------------------- small-N bf16 WMMA GEMM (Bw has 16 rows)
template<int ACT>
__global__ __launch_bounds__(128)
void gemm_small(const bf16_t* __restrict__ A, const bf16_t* __restrict__ Bw,
                const float* __restrict__ bias, float* __restrict__ Cf,
                int M, int N, int K)
{
  const int lane = threadIdx.x & 31;
  const int wave = threadIdx.x >> 5;
  const int half = lane >> 4;
  const int l15  = lane & 15;
  const int m0 = (blockIdx.x * 4 + wave) * 16;

  v8f acc = {};
  const bf16_t* ap = A  + (size_t)(m0 + l15) * K + half * 16;
  const bf16_t* bp = Bw + (size_t)l15 * K + half * 16;   // rows >= N hold junk; columns are independent
  for (int k0 = 0; k0 < K; k0 += 32) {
    v16bf af = join16(*(const v8bf*)(ap + k0), *(const v8bf*)(ap + k0 + 8));
    v16bf bf = join16(*(const v8bf*)(bp + k0), *(const v8bf*)(bp + k0 + 8));
    acc = __builtin_amdgcn_wmma_f32_16x16x32_bf16(false, af, false, bf, (short)0, acc, false, false);
  }
  if (l15 >= N) return;
#pragma unroll
  for (int r = 0; r < 8; ++r) {
    int mm = m0 + r + half * 8;
    float v = acc[r] + bias[l15];
    if constexpr (ACT == ACT_SIGMOID) v = sigm(v);
    Cf[(size_t)mm * N + l15] = v;
  }
}

// ---------------------------------------------------------------- implicit-GEMM 3x3 ring conv
// In: (bs,128,130,256) bf16 halo channel-last; Wr: (co, tap, ci) bf16; Y: (bs,128,128,256) f32
// B panel (16 co x 256 ci per tap) staged in LDS via TDM, double buffered.
__global__ __launch_bounds__(128)
void conv_gemm(const bf16_t* __restrict__ In, const bf16_t* __restrict__ Wr,
               const float* __restrict__ bias, float* __restrict__ Y)
{
  __shared__ bf16_t bt[2][16*256];
  const int lane = threadIdx.x & 31;
  const int wave = threadIdx.x >> 5;
  const int half = lane >> 4;
  const int l15  = lane & 15;
  const int m0 = (blockIdx.x * 4 + wave) * 16;
  const int n0 = blockIdx.y * 16;
  const int b    = m0 >> 14;
  const int phi  = (m0 >> 7) & 127;
  const int rho0 = (m0 & 127);
  const bf16_t* wbase = Wr + (size_t)n0 * (9*256);

#if USE_TDM
  if (wave == 0) {
    tdm_load_wtile(wbase, (unsigned)(uintptr_t)&bt[0][0]);
    __builtin_amdgcn_s_wait_tensorcnt(0);
  }
#else
  for (int i = threadIdx.x; i < 16*32; i += 128) {   // 16 rows x 32 v8bf
    int row = i >> 5, col = (i & 31) * 8;
    *(v8bf*)(&bt[0][row*256 + col]) = *(const v8bf*)(wbase + (size_t)row*2304 + col);
  }
#endif
  __syncthreads();

  v8f acc = {};
  for (int tap = 0; tap < 9; ++tap) {
    const int cur = tap & 1;
#if USE_TDM
    if (wave == 0 && tap < 8)
      tdm_load_wtile(wbase + (tap+1)*256, (unsigned)(uintptr_t)&bt[(tap+1)&1][0]);
#else
    if (tap < 8) {
      for (int i = threadIdx.x; i < 16*32; i += 128) {
        int row = i >> 5, col = (i & 31) * 8;
        *(v8bf*)(&bt[(tap+1)&1][row*256 + col]) =
            *(const v8bf*)(wbase + (tap+1)*256 + (size_t)row*2304 + col);
      }
    }
#endif
    const int kh = tap / 3, kw = tap % 3;
    const int phis = (phi + kh + 127) & 127;                 // ring pad in phi
    const bf16_t* ap = In + (((size_t)(b*128 + phis)) * RHO_H + rho0 + l15 + kw) * 256 + half * 16;
    const bf16_t* bp = &bt[cur][l15*256 + half*16];
    for (int k0 = 0; k0 < 256; k0 += 32) {
      v16bf af = join16(*(const v8bf*)(ap + k0), *(const v8bf*)(ap + k0 + 8));
      v16bf bf = join16(*(const v8bf*)(bp + k0), *(const v8bf*)(bp + k0 + 8));
      acc = __builtin_amdgcn_wmma_f32_16x16x32_bf16(false, af, false, bf, (short)0, acc, false, false);
    }
#if USE_TDM
    if (wave == 0 && tap < 8) __builtin_amdgcn_s_wait_tensorcnt(0);
#endif
    __syncthreads();
  }
#pragma unroll
  for (int r = 0; r < 8; ++r) {
    int mm = m0 + r + half * 8;
    Y[(size_t)mm * 256 + n0 + l15] = acc[r] + bias[n0 + l15];
  }
}

// ---------------------------------------------------------------- batchnorm (over bs*128*128)
__global__ __launch_bounds__(256)
void bn_reduce(const float* __restrict__ Y, float* __restrict__ gsum, float* __restrict__ gsq){
  int c = threadIdx.x;
  int row0 = blockIdx.x * 128;
  float s = 0.f, s2 = 0.f;
  for (int r = 0; r < 128; ++r) {
    float v = Y[(size_t)(row0 + r) * 256 + c];
    s += v; s2 += v * v;
  }
  atomicAdd(&gsum[c], s);
  atomicAdd(&gsq[c], s2);
}
__global__ __launch_bounds__(256)
void bn_finalize(const float* __restrict__ gsum, const float* __restrict__ gsq,
                 const float* __restrict__ g, const float* __restrict__ be,
                 float* __restrict__ scale, float* __restrict__ shift){
  int c = threadIdx.x;
  float mu  = gsum[c] * (1.f / 32768.f);
  float var = gsq[c] * (1.f / 32768.f) - mu * mu;
  float sc  = g[c] * rsqrtf(var + 1e-5f);
  scale[c] = sc;
  shift[c] = be[c] - mu * sc;
}
template<bool WRITE_OUT>
__global__ __launch_bounds__(256)
void bn_apply(const float* __restrict__ Y, const float* __restrict__ scale,
              const float* __restrict__ shift, bf16_t* __restrict__ nextIn,
              float* __restrict__ outF, bf16_t* __restrict__ outB){
  int m = blockIdx.x, c = threadIdx.x;
  size_t idx = (size_t)m * 256 + c;
  float v = fmaxf(Y[idx] * scale[c] + shift[c], 0.f);
  nextIn[((size_t)(m >> 7) * RHO_H + (m & 127) + 1) * 256 + c] = (bf16_t)v;   // halo layout
  if constexpr (WRITE_OUT) {
    int q = m & (Q_N - 1), b = m >> 14;
    size_t mo = (size_t)(q * 2 + b) * 256 + c;
    outF[mo] = v;
    outB[mo] = (bf16_t)v;
  }
}

// ---------------------------------------------------------------- per-row layernorm(+relu) -> bf16
template<bool DO_LN>
__global__ __launch_bounds__(256)
void ln_relu(const float* __restrict__ T, const float* __restrict__ g,
             const float* __restrict__ be, bf16_t* __restrict__ Out){
  __shared__ float s1[256], s2[256];
  int m = blockIdx.x, c = threadIdx.x;
  float v = T[(size_t)m * 256 + c];
  float o;
  if constexpr (DO_LN) {
    s1[c] = v; s2[c] = v * v;
    __syncthreads();
    for (int s = 128; s > 0; s >>= 1) {
      if (c < s) { s1[c] += s1[c + s]; s2[c] += s2[c + s]; }
      __syncthreads();
    }
    float mu  = s1[0] * (1.f / 256.f);
    float var = s2[0] * (1.f / 256.f) - mu * mu;
    o = (v - mu) * rsqrtf(var + 1e-5f) * g[c] + be[c];
  } else {
    o = v;
  }
  Out[(size_t)m * 256 + c] = (bf16_t)fmaxf(o, 0.f);
}

// ---------------------------------------------------------------- zz update + regs output
__global__ void zz_update(const float* __restrict__ tmp, float* __restrict__ zzb,
                          float* __restrict__ regs){
  int m = blockIdx.x * blockDim.x + threadIdx.x;
  if (m >= M_N) return;
  int q = m >> 1, b = m & 1;
  float v = sigm(inv_sigm(zzb[b * Q_N + q]) + tmp[m]);
  zzb[b * Q_N + q]  = v;
  regs[b * Q_N + q] = v;
}

// ================================================================ host
extern "C" void kernel_launch(void* const* d_in, const int* in_sizes, int n_in,
                              void* d_out, int out_size, void* d_ws, size_t ws_size,
                              hipStream_t stream) {
  const float* x_in   = (const float*)d_in[0];
  const float* l2i    = (const float*)d_in[1];
  const float* zz     = (const float*)d_in[4];
  const float* phi_q  = (const float*)d_in[5];
  const float* rho_q  = (const float*)d_in[6];
  const float* ref_W  = (const float*)d_in[7];
  const float* ref_b  = (const float*)d_in[8];
  const float* aw_W   = (const float*)d_in[9];
  const float* aw_b   = (const float*)d_in[10];
  const float* projW  = (const float*)d_in[11];
  const float* projb  = (const float*)d_in[12];
  const float* c1W    = (const float*)d_in[13];
  const float* c1b    = (const float*)d_in[14];
  const float* bn1g   = (const float*)d_in[15];
  const float* bn1b   = (const float*)d_in[16];
  const float* c2W    = (const float*)d_in[17];
  const float* c2b    = (const float*)d_in[18];
  const float* bn2g   = (const float*)d_in[19];
  const float* bn2b   = (const float*)d_in[20];
  const float* regW1  = (const float*)d_in[21];
  const float* regb1  = (const float*)d_in[22];
  const float* regW2  = (const float*)d_in[23];
  const float* regb2  = (const float*)d_in[24];
  const float* segW1  = (const float*)d_in[25];
  const float* segb1  = (const float*)d_in[26];
  const float* segg   = (const float*)d_in[27];
  const float* segbe  = (const float*)d_in[28];
  const float* segW2  = (const float*)d_in[29];
  const float* segb2  = (const float*)d_in[30];
  const float* offW1  = (const float*)d_in[31];
  const float* offb1  = (const float*)d_in[32];
  const float* offg   = (const float*)d_in[33];
  const float* offbe  = (const float*)d_in[34];
  const float* offW2  = (const float*)d_in[35];
  const float* offb2  = (const float*)d_in[36];
  const float* cenW1  = (const float*)d_in[37];
  const float* cenb1  = (const float*)d_in[38];
  const float* ceng   = (const float*)d_in[39];
  const float* cenbe  = (const float*)d_in[40];
  const float* cenW2  = (const float*)d_in[41];
  const float* cenb2  = (const float*)d_in[42];
  (void)in_sizes; (void)n_in; (void)out_size; (void)ws_size;

  // d_out layout (floats): output | segs | offs | cens | regs
  float* outF = (float*)d_out;                 // (Q, bs, E)
  float* segs = outF + (size_t)M_N * 256;      // (L, Q, bs, 4)
  float* offs = segs + (size_t)2 * M_N * 4;    // (L, Q, bs, 2)
  float* cens = offs + (size_t)2 * M_N * 2;    // (L, Q, bs, 1)
  float* regs = cens + (size_t)2 * M_N;        // (L, bs, Q, 1)

  // workspace carve-up
  uint8_t* ws = (uint8_t*)d_ws;
  size_t off = 0;
  auto carve = [&](size_t bytes) { size_t r = off; off = (off + bytes + 255) & ~(size_t)255; return r; };
  float*  qe      = (float*)(ws + carve((size_t)M_N * 256 * 4));
  float*  zzb     = (float*)(ws + carve((size_t)M_N * 4));
  bf16_t* aggB    = (bf16_t*)(ws + carve((size_t)M_N * 256 * 2));      // also head t-bf16
  bf16_t* convin1 = (bf16_t*)(ws + carve((size_t)256 * RHO_H * 256 * 2)); // (bs*128, 130, 256)
  bf16_t* convin2 = (bf16_t*)(ws + carve((size_t)256 * RHO_H * 256 * 2));
  float*  ybuf    = (float*)(ws + carve((size_t)M_N * 256 * 4));       // conv out, head t
  bf16_t* outB    = (bf16_t*)(ws + carve((size_t)M_N * 256 * 2));
  float*  tmpv    = (float*)(ws + carve((size_t)M_N * 4));
  float*  bnstat  = (float*)(ws + carve(4 * 256 * 4));                 // gsum, gsq, scale, shift
  bf16_t* wConv   = (bf16_t*)(ws + carve((size_t)256 * 2304 * 2));
  bf16_t* wB      = (bf16_t*)(ws + carve((size_t)256 * 256 * 2));
  bf16_t* wC      = (bf16_t*)(ws + carve((size_t)16 * 256 * 2));

  dim3 gridWide(M_N / 64, 4);          // N = 256
  dim3 gridConv(M_N / 64, 16);
  dim3 gridSmall(M_N / 64);

  halo_zero<<<512, 256, 0, stream>>>(convin1);
  halo_zero<<<512, 256, 0, stream>>>(convin2);
  init_kernel<<<Q_N, 256, 0, stream>>>(phi_q, rho_q, ref_W, ref_b, zz, qe, outF, zzb);

  for (int l = 0; l < 2; ++l) {
    // --- fused attention + feature sampling + camera aggregation
    sample_agg<<<M_N, 256, 0, stream>>>(x_in, l2i, outF, qe, zzb,
                                        aw_W + l*24*256, aw_b + l*24, aggB);
    // --- proj GEMM: convin1 = agg @ projW^T + projb + output (halo channel-last bf16)
    f32_to_bf16<<<256, 256, 0, stream>>>(projW + l*65536, wB, 65536);
    gemm_wide<true, false, true><<<gridWide, 128, 0, stream>>>(
        aggB, wB, projb + l*256, outF, nullptr, convin1, M_N, 256, 256);
    // --- ring conv block 1
    conv_w_repack<<<2304, 256, 0, stream>>>(c1W + (size_t)l*589824, wConv);
    conv_gemm<<<gridConv, 128, 0, stream>>>(convin1, wConv, c1b + l*256, ybuf);
    zero_f32<<<2, 256, 0, stream>>>(bnstat, 512);
    bn_reduce<<<256, 256, 0, stream>>>(ybuf, bnstat, bnstat + 256);
    bn_finalize<<<1, 256, 0, stream>>>(bnstat, bnstat + 256, bn1g + l*256, bn1b + l*256,
                                       bnstat + 512, bnstat + 768);
    bn_apply<false><<<M_N, 256, 0, stream>>>(ybuf, bnstat + 512, bnstat + 768,
                                             convin2, nullptr, nullptr);
    // --- ring conv block 2 (writes new output + bf16 copy)
    conv_w_repack<<<2304, 256, 0, stream>>>(c2W + (size_t)l*589824, wConv);
    conv_gemm<<<gridConv, 128, 0, stream>>>(convin2, wConv, c2b + l*256, ybuf);
    zero_f32<<<2, 256, 0, stream>>>(bnstat, 512);
    bn_reduce<<<256, 256, 0, stream>>>(ybuf, bnstat, bnstat + 256);
    bn_finalize<<<1, 256, 0, stream>>>(bnstat, bnstat + 256, bn2g + l*256, bn2b + l*256,
                                       bnstat + 512, bnstat + 768);
    bn_apply<true><<<M_N, 256, 0, stream>>>(ybuf, bnstat + 512, bnstat + 768,
                                            convin1, outF, outB);
    // --- reg head: relu(out@W1+b1)@W2+b2 -> zz update
    f32_to_bf16<<<256, 256, 0, stream>>>(regW1 + l*65536, wB, 65536);
    gemm_wide<false, true, false><<<gridWide, 128, 0, stream>>>(
        outB, wB, regb1 + l*256, nullptr, ybuf, nullptr, M_N, 256, 256);
    ln_relu<false><<<M_N, 256, 0, stream>>>(ybuf, nullptr, nullptr, aggB);
    f32_to_bf16<<<1, 256, 0, stream>>>(regW2 + l*256, wC, 256);
    gemm_small<ACT_NONE><<<gridSmall, 128, 0, stream>>>(aggB, wC, regb2 + l, tmpv, M_N, 1, 256);
    zz_update<<<128, 256, 0, stream>>>(tmpv, zzb, regs + (size_t)l * M_N);
    // --- seg head (N=4)
    f32_to_bf16<<<256, 256, 0, stream>>>(segW1 + l*65536, wB, 65536);
    gemm_wide<false, true, false><<<gridWide, 128, 0, stream>>>(
        outB, wB, segb1 + l*256, nullptr, ybuf, nullptr, M_N, 256, 256);
    ln_relu<true><<<M_N, 256, 0, stream>>>(ybuf, segg + l*256, segbe + l*256, aggB);
    f32_to_bf16<<<4, 256, 0, stream>>>(segW2 + l*1024, wC, 1024);
    gemm_small<ACT_NONE><<<gridSmall, 128, 0, stream>>>(aggB, wC, segb2 + l*4,
                                                        segs + (size_t)l * M_N * 4, M_N, 4, 256);
    // --- off head (N=2)
    f32_to_bf16<<<256, 256, 0, stream>>>(offW1 + l*65536, wB, 65536);
    gemm_wide<false, true, false><<<gridWide, 128, 0, stream>>>(
        outB, wB, offb1 + l*256, nullptr, ybuf, nullptr, M_N, 256, 256);
    ln_relu<true><<<M_N, 256, 0, stream>>>(ybuf, offg + l*256, offbe + l*256, aggB);
    f32_to_bf16<<<2, 256, 0, stream>>>(offW2 + l*512, wC, 512);
    gemm_small<ACT_NONE><<<gridSmall, 128, 0, stream>>>(aggB, wC, offb2 + l*2,
                                                        offs + (size_t)l * M_N * 2, M_N, 2, 256);
    // --- cen head (N=1, sigmoid)
    f32_to_bf16<<<256, 256, 0, stream>>>(cenW1 + l*65536, wB, 65536);
    gemm_wide<false, true, false><<<gridWide, 128, 0, stream>>>(
        outB, wB, cenb1 + l*256, nullptr, ybuf, nullptr, M_N, 256, 256);
    ln_relu<true><<<M_N, 256, 0, stream>>>(ybuf, ceng + l*256, cenbe + l*256, aggB);
    f32_to_bf16<<<1, 256, 0, stream>>>(cenW2 + l*256, wC, 256);
    gemm_small<ACT_SIGMOID><<<gridSmall, 128, 0, stream>>>(aggB, wC, cenb2 + l,
                                                           cens + (size_t)l * M_N, M_N, 1, 256);
  }
}
